// ResBlock_52458730553893
// MI455X (gfx1250) — matrix operand order repaired and checked by reading, not compiled, not run on hardware
//
#include <hip/hip_runtime.h>

// ---------------------------------------------------------------------------
// MI455X (gfx1250) sparse-conv ResBlock:
//   h   = relu(bn(gather(x) @ W1))   K=27*64   v_wmma_f32_16x16x32_f16
//   h   = relu(bn(gather(h) @ W2))   K=27*128  v_wmma_f32_16x16x32_f16
//   out = h + bn(x @ Wd)             K=64      v_wmma_f32_16x16x32_f16
// fp32 inputs -> f16 operands, fp32 accumulation.
//  * B staged via double-buffered LDS (16KB/stage, shared by 8 waves)
//  * K-step = 64: 16 WMMAs per stage, one barrier per stage
//  * gathered A fragments register double-buffered (1-stage lead)
//  * neighbor indices prefetched with 2-stage lead
//  * sched_group_barrier pipelines DS-reads 2 tiles ahead of each WMMA
// ---------------------------------------------------------------------------

typedef __attribute__((ext_vector_type(16))) _Float16     v16h;
typedef __attribute__((ext_vector_type(8)))  float        v8f;
typedef __attribute__((ext_vector_type(4)))  unsigned int u32x4;
typedef __attribute__((ext_vector_type(4)))  float        f32x4;
typedef __attribute__((ext_vector_type(4)))  _Float16     h16x4;

union ABFrag { u32x4 u[2]; v16h h; };

#if __has_builtin(__builtin_amdgcn_sched_group_barrier)
#define SGB(mask, size) __builtin_amdgcn_sched_group_barrier(mask, size, 0)
#else
#define SGB(mask, size)
#endif

// Pipeline shape for one K=64 stage:
//   VMEM reads first, 2 tiles of DS reads up front, then alternate
//   (2 WMMAs / 4 DS reads) so data is fetched 2 tiles ahead; commits last.
__device__ __forceinline__ void sched_pipeline() {
  SGB(0x020, 9);                 // global: 4 B-stage + 4 A-prefetch + 1 index
  SGB(0x100, 8);                 // DS read: B fragments, tiles 0-1
  SGB(0x008, 2); SGB(0x100, 4);  // WMMA t0 / DS t2
  SGB(0x008, 2); SGB(0x100, 4);  // WMMA t1 / DS t3
  SGB(0x008, 2); SGB(0x100, 4);
  SGB(0x008, 2); SGB(0x100, 4);
  SGB(0x008, 2); SGB(0x100, 4);
  SGB(0x008, 2); SGB(0x100, 4);  // WMMA t5 / DS t7
  SGB(0x008, 4);                 // WMMA t6, t7
  SGB(0x200, 4);                 // DS write: commit next stage
}

// ---------------- elementwise converts (128-bit accesses) -------------------
__global__ __launch_bounds__(256) void cvt_f32_to_f16(
    const float* __restrict__ src, _Float16* __restrict__ dst, long total4) {
  for (long i = (long)blockIdx.x * blockDim.x + threadIdx.x; i < total4;
       i += (long)gridDim.x * blockDim.x) {
    f32x4 v = ((const f32x4*)src)[i];
    h16x4 h;
#pragma unroll
    for (int j = 0; j < 4; ++j) h[j] = (_Float16)v[j];
    ((h16x4*)dst)[i] = h;
  }
}

// W [KT, 128] fp32 (row-major)  ->  Wt [128][KT] f16 (B-matrix friendly)
__global__ __launch_bounds__(256) void wtranspose(
    const float* __restrict__ W, _Float16* __restrict__ Wt, long KT) {
  long total = KT * 128;
  for (long i = (long)blockIdx.x * blockDim.x + threadIdx.x; i < total;
       i += (long)gridDim.x * blockDim.x) {
    long d  = i & 127;
    long kc = i >> 7;
    Wt[d * KT + kc] = (_Float16)W[kc * 128 + d];
  }
}

// ---------------- WMMA GEMM with on-the-fly neighbor gather ----------------
// X  : [N, CIN] f16 features
// nbr: [N, KNBR] int32 (ignored when KNBR==1 -> 1x1 conv / identity)
// Wt : [128][KNBR*CIN] f16 (output-channel-major, K contiguous)
// Y  : [N, 128] fp32
template <int CIN, int KNBR>
__global__ __launch_bounds__(256) void gemm_wmma_gather(
    const _Float16* __restrict__ X, const int* __restrict__ nbr,
    const _Float16* __restrict__ Wt, float* __restrict__ Y, int N) {
  constexpr int KT     = KNBR * CIN;
  constexpr int KSTEPS = KT / 64;   // K=64 per stage

  // double-buffered B stage: 128 d-rows x 64 k halves = 16KB per buffer
  __shared__ _Float16 lds[2][128 * 64];

  const int tid  = threadIdx.x;
  const int lane = tid & 31;
  const int wave = tid >> 5;
  const int m    = lane & 15;  // A row / B col / D col within 16x16 tile
  const int hi   = lane >> 4;  // lane half selects K sub-range

  // B staging role: thread tid loads 64B of weight row d = tid/2
  const int sd = tid >> 1;
  const int sp = tid & 1;
  const _Float16* sgp = Wt + (long)sd * KT + sp * 32;
  const int       sloff = sd * 64 + sp * 32;  // halfword offset within buffer

  const long row0 = (long)blockIdx.x * 128 + wave * 16;
  long r = row0 + m;
  if (r >= N) r = N - 1;  // clamp gather rows; stores predicated below

  v8f acc[8];
#pragma unroll
  for (int t = 0; t < 8; ++t) acc[t] = {};

  // ---- prologue: stage B(0) into buf0; A(0) data; index for A(1) ----
  {
    u32x4 p0 = *(const u32x4*)(sgp);
    u32x4 p1 = *(const u32x4*)(sgp + 8);
    u32x4 p2 = *(const u32x4*)(sgp + 16);
    u32x4 p3 = *(const u32x4*)(sgp + 24);
    _Float16* dst = &lds[0][0] + sloff;
    *(u32x4*)(dst)      = p0;
    *(u32x4*)(dst + 8)  = p1;
    *(u32x4*)(dst + 16) = p2;
    *(u32x4*)(dst + 24) = p3;
  }

  ABFrag a0, a1, n0, n1;  // current / next gathered A fragments (K=64)
  {
    const long nrow = (KNBR == 1) ? r : (long)nbr[r * KNBR];
    const _Float16* ap = X + nrow * CIN + hi * 8;
    a0.u[0] = *(const u32x4*)(ap);
    a0.u[1] = *(const u32x4*)(ap + 16);
    a1.u[0] = *(const u32x4*)(ap + 32);
    a1.u[1] = *(const u32x4*)(ap + 48);
  }
  long idx_nxt = r;  // neighbor row for stage ks+1
  if (KNBR != 1 && KSTEPS > 1) idx_nxt = (long)nbr[r * KNBR + 64 / CIN];

  __syncthreads();

  u32x4 s0, s1, s2, s3;
  for (int ks = 0; ks < KSTEPS; ++ks) {
    const int cur = ks & 1;

    // stage B(ks+1) into registers; prefetch A(ks+1) via 1-stage-old index
    if (ks + 1 < KSTEPS) {
      const int Kb1 = (ks + 1) * 64;
      s0 = *(const u32x4*)(sgp + Kb1);
      s1 = *(const u32x4*)(sgp + Kb1 + 8);
      s2 = *(const u32x4*)(sgp + Kb1 + 16);
      s3 = *(const u32x4*)(sgp + Kb1 + 24);
      const int chan = Kb1 % CIN;
      const _Float16* ap = X + idx_nxt * CIN + chan + hi * 8;
      n0.u[0] = *(const u32x4*)(ap);
      n0.u[1] = *(const u32x4*)(ap + 16);
      n1.u[0] = *(const u32x4*)(ap + 32);
      n1.u[1] = *(const u32x4*)(ap + 48);
    }
    // prefetch neighbor index for stage ks+2 (2-stage lead)
    long idx_nn = r;
    if (KNBR != 1 && ks + 2 < KSTEPS)
      idx_nn = (long)nbr[r * KNBR + ((ks + 2) * 64) / CIN];

    // 8 col-tiles x (2 DS fragment pairs + 2 WMMAs); sched_group_barrier
    // below rearranges so DS reads run 2 tiles ahead of the WMMAs.
    const _Float16* lb = &lds[cur][0];
#pragma unroll
    for (int t = 0; t < 8; ++t) {
      const _Float16* bp = lb + (t * 16 + m) * 64 + hi * 8;
      ABFrag b0, b1;
      b0.u[0] = *(const u32x4*)(bp);
      b0.u[1] = *(const u32x4*)(bp + 16);
      b1.u[0] = *(const u32x4*)(bp + 32);
      b1.u[1] = *(const u32x4*)(bp + 48);
      acc[t] = __builtin_amdgcn_wmma_f32_16x16x32_f16(
          false, a0.h, false, b0.h, (short)0, acc[t], false, false);
      acc[t] = __builtin_amdgcn_wmma_f32_16x16x32_f16(
          false, a1.h, false, b1.h, (short)0, acc[t], false, false);
    }

    // commit staged B for ks+1 into the other buffer; rotate A + index
    if (ks + 1 < KSTEPS) {
      _Float16* dst = &lds[1 - cur][0] + sloff;
      *(u32x4*)(dst)      = s0;
      *(u32x4*)(dst + 8)  = s1;
      *(u32x4*)(dst + 16) = s2;
      *(u32x4*)(dst + 24) = s3;
      a0 = n0;
      a1 = n1;
      idx_nxt = idx_nn;
    }
    sched_pipeline();
    __syncthreads();
  }

  // D layout: VGPR g, lanes 0-15 -> M=g, lanes 16-31 -> M=8+g; N = lane%16
#pragma unroll
  for (int t = 0; t < 8; ++t) {
#pragma unroll
    for (int g = 0; g < 8; ++g) {
      long rr = row0 + hi * 8 + g;
      if (rr < N) Y[rr * 128 + t * 16 + m] = acc[t][g];
    }
  }
}

// ---------------- BatchNorm statistics (deterministic two-stage) -----------
__global__ __launch_bounds__(128) void colstats(
    const float* __restrict__ Y, float* __restrict__ partS,
    float* __restrict__ partQ, int N) {
  const int c = threadIdx.x;  // 128 channels
  float s = 0.f, q = 0.f;
  for (long r = blockIdx.x; r < N; r += gridDim.x) {
    float v = Y[r * 128 + c];
    s += v;
    q += v * v;
  }
  partS[(long)blockIdx.x * 128 + c] = s;
  partQ[(long)blockIdx.x * 128 + c] = q;
}

__global__ __launch_bounds__(128) void bn_finalize(
    const float* __restrict__ partS, const float* __restrict__ partQ,
    const float* __restrict__ gamma, const float* __restrict__ beta,
    float* __restrict__ scale, float* __restrict__ shift, int nblocks, int N) {
  const int c = threadIdx.x;
  float s = 0.f, q = 0.f;
  for (int b = 0; b < nblocks; ++b) {
    s += partS[(long)b * 128 + c];
    q += partQ[(long)b * 128 + c];
  }
  const float mu  = s / (float)N;
  const float var = q / (float)N - mu * mu;
  const float inv = rsqrtf(var + 1e-5f);
  const float sc  = gamma[c] * inv;
  scale[c] = sc;
  shift[c] = beta[c] - mu * sc;
}

// h = (f16) relu(y*scale + shift)   (f16 input for conv2); 4 elems/thread
__global__ __launch_bounds__(256) void apply_bn_relu_f16(
    const float* __restrict__ Y, const float* __restrict__ scale,
    const float* __restrict__ shift, _Float16* __restrict__ H, long total4) {
  for (long i = (long)blockIdx.x * blockDim.x + threadIdx.x; i < total4;
       i += (long)gridDim.x * blockDim.x) {
    const int c0 = (int)(i & 31) * 4;  // row length 128 = 32 quads
    f32x4 v  = ((const f32x4*)Y)[i];
    f32x4 sc = *(const f32x4*)(scale + c0);
    f32x4 sh = *(const f32x4*)(shift + c0);
    h16x4 h;
#pragma unroll
    for (int j = 0; j < 4; ++j)
      h[j] = (_Float16)fmaxf(fmaf(v[j], sc[j], sh[j]), 0.f);
    ((h16x4*)H)[i] = h;
  }
}

// out = relu(y2*s2 + t2) + (d*sd + td)   (in-place on y2 == d_out)
__global__ __launch_bounds__(256) void final_add(
    float* __restrict__ Y2, const float* __restrict__ D,
    const float* __restrict__ s2, const float* __restrict__ t2,
    const float* __restrict__ sd, const float* __restrict__ td, long total4) {
  for (long i = (long)blockIdx.x * blockDim.x + threadIdx.x; i < total4;
       i += (long)gridDim.x * blockDim.x) {
    const int c0 = (int)(i & 31) * 4;
    f32x4 y = ((const f32x4*)Y2)[i];
    f32x4 d = ((const f32x4*)D)[i];
    f32x4 a2 = *(const f32x4*)(s2 + c0), b2 = *(const f32x4*)(t2 + c0);
    f32x4 ad = *(const f32x4*)(sd + c0), bd = *(const f32x4*)(td + c0);
    f32x4 o;
#pragma unroll
    for (int j = 0; j < 4; ++j)
      o[j] = fmaxf(fmaf(y[j], a2[j], b2[j]), 0.f) + fmaf(d[j], ad[j], bd[j]);
    ((f32x4*)Y2)[i] = o;
  }
}

// ---------------------------------------------------------------------------
extern "C" void kernel_launch(void* const* d_in, const int* in_sizes, int n_in,
                              void* d_out, int out_size, void* d_ws,
                              size_t ws_size, hipStream_t stream) {
  const float* x   = (const float*)d_in[0];   // [N,64]
  const int*   nbr = (const int*)d_in[1];     // [N,27]
  const float* W1  = (const float*)d_in[2];   // [27,64,128]
  const float* g1  = (const float*)d_in[3];
  const float* b1  = (const float*)d_in[4];
  const float* W2  = (const float*)d_in[5];   // [27,128,128]
  const float* g2  = (const float*)d_in[6];
  const float* b2  = (const float*)d_in[7];
  const float* Wd  = (const float*)d_in[8];   // [64,128]
  const float* gd  = (const float*)d_in[9];
  const float* bd  = (const float*)d_in[10];
  float* out = (float*)d_out;                 // [N,128]

  const int  N   = in_sizes[0] / 64;
  const long KT1 = 27 * 64, KT2 = 27 * 128, KTD = 64;

  // ---- carve workspace (256B aligned regions) ----
  char*  ws = (char*)d_ws;
  size_t o  = 0;
  auto take = [&](size_t bytes) {
    size_t r = o;
    o += (bytes + 255) & ~(size_t)255;
    return r;
  };
  _Float16* xh  = (_Float16*)(ws + take((size_t)N * 64 * 2));
  _Float16* W1t = (_Float16*)(ws + take((size_t)KT1 * 128 * 2));
  _Float16* W2t = (_Float16*)(ws + take((size_t)KT2 * 128 * 2));
  _Float16* Wdt = (_Float16*)(ws + take((size_t)KTD * 128 * 2));
  float*    y1  = (float*)(ws + take((size_t)N * 128 * 4)); // conv1 out; reused as downsample out
  _Float16* h1h = (_Float16*)(ws + take((size_t)N * 128 * 2));
  const int STAT_BLOCKS = 256;
  float* partS  = (float*)(ws + take((size_t)STAT_BLOCKS * 128 * 4));
  float* partQ  = (float*)(ws + take((size_t)STAT_BLOCKS * 128 * 4));
  float* scale1 = (float*)(ws + take(128 * 4));
  float* shift1 = (float*)(ws + take(128 * 4));
  float* scale2 = (float*)(ws + take(128 * 4));
  float* shift2 = (float*)(ws + take(128 * 4));
  float* scaleD = (float*)(ws + take(128 * 4));
  float* shiftD = (float*)(ws + take(128 * 4));

  const dim3 eb(256);
  const int  gemmBlocks = (N + 127) / 128;
  const long NE4 = (long)N * 128 / 4;

  // ---- precision / layout prep ----
  cvt_f32_to_f16<<<dim3(1024), eb, 0, stream>>>(x, xh, (long)N * 64 / 4);
  wtranspose<<<dim3(1024), eb, 0, stream>>>(W1, W1t, KT1);
  wtranspose<<<dim3(1024), eb, 0, stream>>>(W2, W2t, KT2);
  wtranspose<<<dim3(64),   eb, 0, stream>>>(Wd, Wdt, KTD);

  // ---- conv1: y1 = gather(x) @ W1 ----
  gemm_wmma_gather<64, 27><<<dim3(gemmBlocks), eb, 0, stream>>>(xh, nbr, W1t, y1, N);
  colstats<<<dim3(STAT_BLOCKS), dim3(128), 0, stream>>>(y1, partS, partQ, N);
  bn_finalize<<<dim3(1), dim3(128), 0, stream>>>(partS, partQ, g1, b1, scale1, shift1, STAT_BLOCKS, N);
  apply_bn_relu_f16<<<dim3(2048), eb, 0, stream>>>(y1, scale1, shift1, h1h, NE4);

  // ---- downsample: d = x @ Wd (reuse y1 buffer) ----
  gemm_wmma_gather<64, 1><<<dim3(gemmBlocks), eb, 0, stream>>>(xh, (const int*)nullptr, Wdt, y1, N);
  colstats<<<dim3(STAT_BLOCKS), dim3(128), 0, stream>>>(y1, partS, partQ, N);
  bn_finalize<<<dim3(1), dim3(128), 0, stream>>>(partS, partQ, gd, bd, scaleD, shiftD, STAT_BLOCKS, N);

  // ---- conv2: y2 = gather(h1) @ W2, written straight into d_out ----
  gemm_wmma_gather<128, 27><<<dim3(gemmBlocks), eb, 0, stream>>>(h1h, nbr, W2t, out, N);
  colstats<<<dim3(STAT_BLOCKS), dim3(128), 0, stream>>>(out, partS, partQ, N);
  bn_finalize<<<dim3(1), dim3(128), 0, stream>>>(partS, partQ, g2, b2, scale2, shift2, STAT_BLOCKS, N);

  // ---- out = relu(bn(y2)) + bn(d) ----
  final_add<<<dim3(2048), eb, 0, stream>>>(out, y1, scale2, shift2, scaleD, shiftD, NE4);

  (void)n_in; (void)out_size; (void)ws_size;
}
